// PMA_24842090840469
// MI455X (gfx1250) — compile-verified, matrix operand order checked by reading, not run on hardware
//
#include <hip/hip_runtime.h>
#include <hip/hip_bf16.h>

// ---------------------------------------------------------------------------
// Multi-hop noisy graph propagation for MI455X (gfx1250, wave32).
//
//   out[0]   = l2norm(x)
//   out[k+1] = l2norm(scatter_add(out[k][src] -> dst) + sigma*noise)
//
// Memory-bound: per hop ~3.3 GB of gather+atomic traffic, but both the source
// and destination [N,128] f32 slices (51.2 MB each) fit in the 192 MB L2, so
// the scatter runs at L2 atomic rate, not HBM rate.  FLOPs are negligible.
//
// CDNA5 specifics used:
//  * global_atomic_add_f32 scatter (1 wave/edge, float4 gather, 4 atomics/lane)
//  * v_wmma_f32_16x16x32_f16 computes the 16-row sum-of-squares reduction for
//    the fused noise+L2-normalize pass: A = (noisy value)^2 in the native
//    16x32 f16 A-matrix lane layout, B = all-ones, D columns = row norms^2.
// ---------------------------------------------------------------------------

typedef __attribute__((ext_vector_type(16))) _Float16 v16h;
typedef __attribute__((ext_vector_type(8)))  float    v8f;

#define D_FEAT 128
#define SIGMA_V 0.1f
#define NORM_EPS 1e-12f

// ---- deterministic counter-based Gaussian (PCG hash + Box-Muller) ----------
__device__ __forceinline__ unsigned pcg_hash(unsigned v) {
    v = v * 747796405u + 2891336453u;
    unsigned w = ((v >> ((v >> 28u) + 4u)) ^ v) * 277803737u;
    return (w >> 22u) ^ w;
}
__device__ __forceinline__ float u01(unsigned h) {
    return (float)(h >> 8) * (1.0f / 16777216.0f) + (0.5f / 16777216.0f);
}
__device__ __forceinline__ float gauss_hash(unsigned seed) {
    float u1 = u01(pcg_hash(seed));
    float u2 = u01(pcg_hash(seed ^ 0x9E3779B9u));
    return __fsqrt_rn(-2.0f * __logf(u1)) * __cosf(6.28318530718f * u2);
}

// ---- zero fill (destination of the scatter) --------------------------------
__global__ void PMA_zero_kernel(float4* __restrict__ p, unsigned n4) {
    unsigned i = blockIdx.x * blockDim.x + threadIdx.x;
    if (i < n4) p[i] = float4{0.f, 0.f, 0.f, 0.f};
}

// ---- edge scatter: one wave per edge ---------------------------------------
// lane l gathers cols [4l,4l+4) of row src as a float4 (coalesced 512B/wave)
// and issues 4 coalesced global_atomic_add_f32 into row dst.
__global__ void PMA_scatter_kernel(const float* __restrict__ src_feat,
                                   float* __restrict__ dst_feat,
                                   const long long* __restrict__ e_src,
                                   const long long* __restrict__ e_dst,
                                   int nedges) {
    const int wave = (blockIdx.x * blockDim.x + threadIdx.x) >> 5;
    const int lane = threadIdx.x & 31;
    if (wave >= nedges) return;
    const long long s = e_src[wave];   // wave-uniform
    const long long d = e_dst[wave];
    const float4 v = *(const float4*)(src_feat + (size_t)s * D_FEAT + lane * 4);
    float* dp = dst_feat + (size_t)d * D_FEAT + lane * 4;
    atomicAdd(dp + 0, v.x);
    atomicAdd(dp + 1, v.y);
    atomicAdd(dp + 2, v.z);
    atomicAdd(dp + 3, v.w);
}

// ---- fused noise + L2 normalize over 16-row tiles via WMMA -----------------
// One wave handles 16 rows x 128 cols.  Data is loaded directly in the CDNA5
// 16x32 f16 A-matrix lane layout:
//   lanes 0-15  (row = lane):    K = {0..7}  and {16..23} of each 32-chunk
//   lanes 16-31 (row = lane-16): K = {8..15} and {24..31}
// A holds squared noisy values; B = ones, so D[m][*] = sum_k A[m][k] = ||row m||^2.
// D layout: lanes 0-15 acc[m] = normsq[m], lanes 16-31 acc[m] = normsq[m+8].
__global__ void PMA_fuse_norm_kernel(const float* __restrict__ src,
                                     float* __restrict__ dst,
                                     int ntiles, unsigned hop_seed, float sigma) {
    const int wave = (blockIdx.x * blockDim.x + threadIdx.x) >> 5;
    if (wave >= ntiles) return;                     // wave-uniform: EXEC stays full
    const int lane = threadIdx.x & 31;
    const int half = lane >> 4;
    const int row  = (wave << 4) + (lane & 15);
    const float* rp = src + (size_t)row * D_FEAT;
    float*       wp = dst + (size_t)row * D_FEAT;

    float vals[64];
    v8f acc = {0.f, 0.f, 0.f, 0.f, 0.f, 0.f, 0.f, 0.f};
    v16h ones;
#pragma unroll
    for (int i = 0; i < 16; ++i) ones[i] = (_Float16)1.0f;

#pragma unroll
    for (int c = 0; c < 4; ++c) {
        const int colbase = c * 32 + half * 8;
        const float4 a0 = *(const float4*)(rp + colbase);
        const float4 a1 = *(const float4*)(rp + colbase + 4);
        const float4 b0 = *(const float4*)(rp + colbase + 16);
        const float4 b1 = *(const float4*)(rp + colbase + 20);
        float tmp[16] = {a0.x, a0.y, a0.z, a0.w, a1.x, a1.y, a1.z, a1.w,
                         b0.x, b0.y, b0.z, b0.w, b1.x, b1.y, b1.z, b1.w};
        v16h A;
#pragma unroll
        for (int i = 0; i < 16; ++i) {
            const int col = colbase + i + ((i >= 8) ? 8 : 0);
            float v = tmp[i];
            if (sigma != 0.0f)
                v += sigma * gauss_hash((unsigned)(row * D_FEAT + col) ^ hop_seed);
            vals[c * 16 + i] = v;
            A[i] = (_Float16)(v * v);
        }
        acc = __builtin_amdgcn_wmma_f32_16x16x32_f16(
            /*neg_a=*/false, A, /*neg_b=*/false, ones,
            /*c_mod=*/(short)0, acc, /*reuse_a=*/false, /*reuse_b=*/false);
    }

    // Each lane needs normsq[rloc], rloc = lane&15.  Local 8-way select gives
    // normsq[half*8 + (rloc&7)]; shuffle from the lane in the correct half.
    const int rloc = lane & 15;
    const int m = rloc & 7;
    float sel = acc[0];
    sel = (m == 1) ? acc[1] : sel;
    sel = (m == 2) ? acc[2] : sel;
    sel = (m == 3) ? acc[3] : sel;
    sel = (m == 4) ? acc[4] : sel;
    sel = (m == 5) ? acc[5] : sel;
    sel = (m == 6) ? acc[6] : sel;
    sel = (m == 7) ? acc[7] : sel;
    const int src_lane = ((rloc >> 3) << 4) | m;
    const float normsq = __shfl(sel, src_lane, 32);
    const float inv = 1.0f / fmaxf(__fsqrt_rn(normsq), NORM_EPS);

#pragma unroll
    for (int c = 0; c < 4; ++c) {
        const int colbase = c * 32 + half * 8;
        const float* v = &vals[c * 16];
        *(float4*)(wp + colbase)      = float4{v[0] * inv,  v[1] * inv,  v[2] * inv,  v[3] * inv};
        *(float4*)(wp + colbase + 4)  = float4{v[4] * inv,  v[5] * inv,  v[6] * inv,  v[7] * inv};
        *(float4*)(wp + colbase + 16) = float4{v[8] * inv,  v[9] * inv,  v[10] * inv, v[11] * inv};
        *(float4*)(wp + colbase + 20) = float4{v[12] * inv, v[13] * inv, v[14] * inv, v[15] * inv};
    }
}

extern "C" void kernel_launch(void* const* d_in, const int* in_sizes, int n_in,
                              void* d_out, int out_size, void* d_ws, size_t ws_size,
                              hipStream_t stream) {
    const float*     x  = (const float*)d_in[0];
    const long long* ei = (const long long*)d_in[1];   // int64 [2, E]
    const int N = in_sizes[0] / D_FEAT;                // 100000
    const int E = in_sizes[1] / 2;                     // 3200000
    float* out = (float*)d_out;                        // [9, N, 128]

    const size_t slice = (size_t)N * D_FEAT;
    const int ntiles = N / 16;                         // 6250 (N % 16 == 0)
    const int norm_blocks = (ntiles * 32 + 255) / 256;
    const unsigned n4 = (unsigned)(slice / 4);
    const int zero_blocks = (int)((n4 + 255) / 256);
    const int scat_blocks = (E * 32 + 255) / 256;      // one wave per edge

    // out[0] = l2norm(x)  (sigma = 0 -> no noise path)
    PMA_fuse_norm_kernel<<<norm_blocks, 256, 0, stream>>>(x, out, ntiles, 0u, 0.0f);

    for (int k = 0; k < 8; ++k) {
        float* prev = out + (size_t)k * slice;
        float* next = out + (size_t)(k + 1) * slice;
        PMA_zero_kernel<<<zero_blocks, 256, 0, stream>>>((float4*)next, n4);
        PMA_scatter_kernel<<<scat_blocks, 256, 0, stream>>>(prev, next, ei, ei + E, E);
        const unsigned hop_seed = 0xA5F1C3D2u + 0x9E3779B9u * (unsigned)k;
        PMA_fuse_norm_kernel<<<norm_blocks, 256, 0, stream>>>(next, next, ntiles,
                                                              hop_seed, SIGMA_V);
    }
}
